// Attention_52467320488225
// MI455X (gfx1250) — matrix operand order repaired and checked by reading, not compiled, not run on hardware
//
#include <hip/hip_runtime.h>
#include <stdint.h>

// ---------- types ----------
typedef __bf16 bf16_t;
typedef __attribute__((ext_vector_type(16))) __bf16 v16bf;
typedef __attribute__((ext_vector_type(8)))  float  v8f;
typedef __attribute__((ext_vector_type(8)))  unsigned int v8u;
typedef __attribute__((ext_vector_type(4)))  unsigned int u32x4;
typedef __attribute__((ext_vector_type(8)))  int i32x8;
typedef __attribute__((ext_vector_type(4)))  int i32x4;

// fp32 -> bf16 round-to-nearest-even
static __device__ __forceinline__ bf16_t f2bf(float f) {
  union { float f; uint32_t u; } v; v.f = f;
  uint32_t r = v.u + 0x7FFFu + ((v.u >> 16) & 1u);
  uint16_t h = (uint16_t)(r >> 16);
  return __builtin_bit_cast(bf16_t, h);
}

// Load a 16x32 bf16 tile in the CDNA5 WMMA 16-bit A-operand layout.
// base points at tile element (0,0); 'stride' is the row stride in elements.
// Lane (half,idx): row = idx, K values = {g*16 + half*8 + p*2 (+1)}.
// Also used for B operands by pointing 'base' at the transposed tile,
// since the B layout mirrors A with rows<->cols.
static __device__ __forceinline__ v16bf load_frag(const bf16_t* base, int stride,
                                                  int half, int idx) {
  v8u u;
  const bf16_t* row = base + (size_t)idx * stride;
#pragma unroll
  for (int g = 0; g < 2; ++g) {
#pragma unroll
    for (int p = 0; p < 4; ++p) {
      int k = g * 16 + half * 8 + p * 2;
      u[g * 4 + p] = *reinterpret_cast<const uint32_t*>(&row[k]);
    }
  }
  return __builtin_bit_cast(v16bf, u);
}

static __device__ __forceinline__ v8f wmma_bf16(v16bf a, v16bf b, v8f c) {
  return __builtin_amdgcn_wmma_f32_16x16x32_bf16(
      /*neg_a=*/false, a, /*neg_b=*/false, b,
      /*c_mod=*/(short)0, c, /*reuse_a=*/false, /*reuse_b=*/false);
}

// ---------- Tensor Data Mover: 2D tile load, global -> LDS ----------
// Builds the D# descriptor per CDNA5 ISA ch.8 and issues TENSOR_LOAD_TO_LDS.
// tensor: rows of 'tensor_w' elements (2B each), row stride 'stride_elems',
// 'tensor_h' rows; tile: tile_w x tile_h starting at gaddr, written to LDS
// at byte offset lds_off, row-major contiguous (tile_w elems per row).
static __device__ __forceinline__ void tdm_load_2d_bf16(
    uint32_t lds_off, const void* gaddr, uint32_t tensor_w, uint32_t tensor_h,
    uint32_t stride_elems, uint32_t tile_w, uint32_t tile_h) {
  uint64_t ga = (uint64_t)(uintptr_t)gaddr;
  u32x4 g0;
  g0[0] = 1u;                                    // count=1, user descriptor
  g0[1] = lds_off;                               // lds_addr (bytes)
  g0[2] = (uint32_t)(ga & 0xFFFFFFFFu);          // global_addr[31:0]
  g0[3] = (uint32_t)((ga >> 32) & 0x01FFFFFFu)   // global_addr[56:32]
          | (2u << 30);                          // type=2 ("image")
  i32x8 g1;
  g1[0] = (int)(1u << 16);                       // wg_mask=0, data_size=1 (2B)
  g1[1] = (int)((tensor_w & 0xFFFFu) << 16);     // tensor_dim0[15:0]
  g1[2] = (int)(((tensor_w >> 16) & 0xFFFFu) |   // tensor_dim0[31:16]
                ((tensor_h & 0xFFFFu) << 16));   // tensor_dim1[15:0]
  g1[3] = (int)(((tensor_h >> 16) & 0xFFFFu) |   // tensor_dim1[31:16]
                ((tile_w & 0xFFFFu) << 16));     // tile_dim0
  g1[4] = (int)(tile_h & 0xFFFFu);               // tile_dim1 (tile_dim2=0)
  g1[5] = (int)stride_elems;                     // tensor_dim0_stride[31:0]
  g1[6] = 0;                                     // stride0[47:32], stride1[15:0]
  g1[7] = 0;                                     // tensor_dim1_stride[47:16]
#if defined(__clang_major__) && (__clang_major__ >= 23)
  __builtin_amdgcn_tensor_load_to_lds(g0, g1, (i32x4)0, (i32x4)0, (i32x8)0, 0);
#else
  __builtin_amdgcn_tensor_load_to_lds(g0, g1, (i32x4)0, (i32x4)0, 0);
#endif
}

// ---------- stage 1: fp32 -> bf16 ----------
__global__ void cvt_f32_bf16(const float* __restrict__ in,
                             bf16_t* __restrict__ out, int n) {
  int i = blockIdx.x * blockDim.x + threadIdx.x;
  if (i < n) out[i] = f2bf(in[i]);
}

// ---------- stage 2/4: bf16 GEMM, C = A(MxK) * B(KxN), row-major ----------
// Block: 256 threads = 8 waves (2x4), block tile 64x128, K-step 32.
// A tile staged via the Tensor Data Mover (wave 0 issues the DMA);
// B tile staged transposed manually (TDM cannot transpose).
template <bool BF16OUT>
__global__ __launch_bounds__(256) void gemm_bf16(const bf16_t* __restrict__ A,
                                                 const bf16_t* __restrict__ B,
                                                 void* __restrict__ Cout,
                                                 int M, int N, int K) {
  constexpr int BM = 64, BN = 128, BK = 32;
  __shared__ bf16_t As[BM][BK];
  __shared__ bf16_t Bt[BN][BK];  // B tile stored transposed: Bt[n][k]

  const int tid  = threadIdx.x;
  const int wave = tid >> 5, lane = tid & 31;
  const int half = lane >> 4, idx = lane & 15;
  const int wm = wave >> 2, wn = wave & 3;  // 2 x 4 wave grid
  const int blockM = blockIdx.x * BM, blockN = blockIdx.y * BN;
  const uint32_t as_lds_off = (uint32_t)(uintptr_t)&As[0][0];

  v8f c00 = {}, c01 = {}, c10 = {}, c11 = {};

  for (int k0 = 0; k0 < K; k0 += BK) {
    // --- stage A tile 64x32 via TDM (wave-uniform branch: single issue) ---
    if (wave == 0) {
      tdm_load_2d_bf16(as_lds_off, &A[(size_t)blockM * K + k0],
                       (uint32_t)K, (uint32_t)M, (uint32_t)K, BK, BM);
    }
    // --- stage B tile transposed: 32x128 -> Bt[128][32] (overlaps DMA) ---
#pragma unroll
    for (int s = 0; s < 2; ++s) {
      int seg = tid * 2 + s;         // 0..511
      int r   = seg >> 4;            // k row 0..31
      int cs  = (seg & 15) * 8;      // n col segment
      uint4 d = *reinterpret_cast<const uint4*>(&B[(size_t)(k0 + r) * N + blockN + cs]);
      const bf16_t* dv = reinterpret_cast<const bf16_t*>(&d);
#pragma unroll
      for (int e = 0; e < 8; ++e) Bt[cs + e][r] = dv[e];
    }
    __builtin_amdgcn_s_wait_tensorcnt(0);   // wave0: DMA complete; others: NOP
    __syncthreads();

    // speculative prefetch of next B tile rows (global_prefetch)
    if (k0 + BK < K)
      __builtin_prefetch(&B[(size_t)(k0 + BK + (tid >> 3)) * N + blockN], 0, 1);

    v16bf a0 = load_frag(&As[wm * 32][0],      BK, half, idx);
    v16bf a1 = load_frag(&As[wm * 32 + 16][0], BK, half, idx);
    v16bf b0 = load_frag(&Bt[wn * 32][0],      BK, half, idx);
    v16bf b1 = load_frag(&Bt[wn * 32 + 16][0], BK, half, idx);

    c00 = wmma_bf16(a0, b0, c00);
    c01 = wmma_bf16(a0, b1, c01);
    c10 = wmma_bf16(a1, b0, c10);
    c11 = wmma_bf16(a1, b1, c11);
    __syncthreads();
  }

  // C layout: lanes 0-15 hold M = v, lanes 16-31 hold M = v + 8; N = idx
  auto storeC = [&](v8f c, int mBase, int nBase) {
#pragma unroll
    for (int v = 0; v < 8; ++v) {
      int m = mBase + v + 8 * half;
      int n = nBase + idx;
      if constexpr (BF16OUT)
        reinterpret_cast<bf16_t*>(Cout)[(size_t)m * N + n] = f2bf(c[v]);
      else
        reinterpret_cast<float*>(Cout)[(size_t)m * N + n] = c[v];
    }
  };
  const int m0 = blockM + wm * 32, n0 = blockN + wn * 32;
  storeC(c00, m0,      n0);
  storeC(c01, m0,      n0 + 16);
  storeC(c10, m0 + 16, n0);
  storeC(c11, m0 + 16, n0 + 16);
}

// ---------- stage 3: flash attention ----------
// Q:  [B*N, 512] bf16, head h at col h*64
// KV: [B*N, 1024] bf16, K at col h*64, V at col 512 + h*64
// O:  [B*N, 512] bf16 (attention output, pre-Wo)
// grid: (B*H, N/128); block 256 = 8 waves; each wave owns 16 query rows.
__global__ __launch_bounds__(256) void flash_attn(const bf16_t* __restrict__ Q,
                                                  const bf16_t* __restrict__ KV,
                                                  bf16_t* __restrict__ O) {
  constexpr int N = 1024, HD = 512, KVLD = 1024, D = 64;
  const int bh = blockIdx.x;
  const int b = bh >> 3, h = bh & 7;
  const int wave = threadIdx.x >> 5, lane = threadIdx.x & 31;
  const int half = lane >> 4, idx = lane & 15;
  const int q0 = blockIdx.y * 128 + wave * 16;

  const bf16_t* Qbase = Q  + ((size_t)(b * N + q0)) * HD + h * D;
  const bf16_t* Kbase = KV + ((size_t)(b * N)) * KVLD + h * D;
  const bf16_t* Vbase = Kbase + HD;  // +512 cols

  __shared__ bf16_t Pbuf[8][16 * 32];   // per-wave P tile (row-major 16x32)
  __shared__ bf16_t Vt[8][D][32];       // per-wave V tile transposed: Vt[d][keyRel]

  // resident Q fragments: 16 rows x 64 d  ->  two K=32 A fragments
  v16bf aq0 = load_frag(Qbase,      HD, half, idx);
  v16bf aq1 = load_frag(Qbase + 32, HD, half, idx);

  v8f o[4] = {};                 // 16 rows x 64 d accumulator (4 subtiles)
  float mrow[8], lrow[8];
#pragma unroll
  for (int v = 0; v < 8; ++v) { mrow[v] = -3.0e38f; lrow[v] = 0.f; }

  for (int j = 0; j < N; j += 32) {
    // ---- S = Q K^T for keys j..j+31 (two 16-key subtiles) ----
    v8f s0 = {}, s1 = {};
    {
      v16bf bk;
      bk = load_frag(Kbase + (size_t)j * KVLD,            KVLD, half, idx);
      s0 = wmma_bf16(aq0, bk, s0);
      bk = load_frag(Kbase + (size_t)j * KVLD + 32,       KVLD, half, idx);
      s0 = wmma_bf16(aq1, bk, s0);
      bk = load_frag(Kbase + (size_t)(j + 16) * KVLD,     KVLD, half, idx);
      s1 = wmma_bf16(aq0, bk, s1);
      bk = load_frag(Kbase + (size_t)(j + 16) * KVLD + 32, KVLD, half, idx);
      s1 = wmma_bf16(aq1, bk, s1);
    }

    // ---- stage V tile transposed while S WMMAs drain ----
    {
      const bf16_t* vrow = Vbase + (size_t)(j + lane) * KVLD;  // token j+lane
#pragma unroll
      for (int dw = 0; dw < 32; ++dw) {
        uint32_t d2 = *reinterpret_cast<const uint32_t*>(&vrow[dw * 2]);
        Vt[wave][dw * 2][lane]     = __builtin_bit_cast(bf16_t, (uint16_t)(d2 & 0xFFFFu));
        Vt[wave][dw * 2 + 1][lane] = __builtin_bit_cast(bf16_t, (uint16_t)(d2 >> 16));
      }
    }

    // ---- online softmax over the 32 new columns ----
    float scale[8];
#pragma unroll
    for (int v = 0; v < 8; ++v) {
      float t = fmaxf(s0[v], s1[v]);
#pragma unroll
      for (int off = 8; off >= 1; off >>= 1) t = fmaxf(t, __shfl_xor(t, off, 16));
      float mn = fmaxf(mrow[v], t);
      scale[v] = __expf(mrow[v] - mn);
      mrow[v] = mn;
      float p0 = __expf(s0[v] - mn);
      float p1 = __expf(s1[v] - mn);
      s0[v] = p0; s1[v] = p1;
      float rs = p0 + p1;
#pragma unroll
      for (int off = 8; off >= 1; off >>= 1) rs += __shfl_xor(rs, off, 16);
      lrow[v] = lrow[v] * scale[v] + rs;
    }
#pragma unroll
    for (int f = 0; f < 4; ++f)
#pragma unroll
      for (int v = 0; v < 8; ++v) o[f][v] *= scale[v];

    // ---- P: C-layout -> row-major LDS -> A-layout fragment ----
#pragma unroll
    for (int v = 0; v < 8; ++v) {
      int r = v + 8 * half;
      Pbuf[wave][r * 32 + idx]      = f2bf(s0[v]);
      Pbuf[wave][r * 32 + 16 + idx] = f2bf(s1[v]);
    }
    v16bf ap = load_frag(&Pbuf[wave][0], 32, half, idx);

    // ---- O += P * V  (V fragments from transposed LDS tile) ----
#pragma unroll
    for (int f = 0; f < 4; ++f) {
      v16bf bv = load_frag(&Vt[wave][f * 16][0], 32, half, idx);
      o[f] = wmma_bf16(ap, bv, o[f]);
    }
  }

  // ---- epilogue: normalize and store bf16 ----
#pragma unroll
  for (int v = 0; v < 8; ++v) {
    float inv = 1.0f / lrow[v];
    int r = q0 + v + 8 * half;
    bf16_t* orow = O + ((size_t)(b * N) + r) * HD + h * D;
#pragma unroll
    for (int f = 0; f < 4; ++f) orow[f * 16 + idx] = f2bf(o[f][v] * inv);
  }
}

// ---------- host launch ----------
extern "C" void kernel_launch(void* const* d_in, const int* in_sizes, int n_in,
                              void* d_out, int out_size, void* d_ws, size_t ws_size,
                              hipStream_t stream) {
  constexpr int B = 8, N = 1024, DM = 512, HD = 512;  // H*D = 512
  constexpr int Mrows = B * N;                        // 8192

  const float* current = (const float*)d_in[0];
  const float* hidden  = (const float*)d_in[1];
  const float* Wq      = (const float*)d_in[2];
  const float* Wkv     = (const float*)d_in[3];
  const float* Wo      = (const float*)d_in[4];
  float* out = (float*)d_out;

  char* ws = (char*)d_ws;
  auto take = [&](size_t nbytes) { char* p = ws; ws += (nbytes + 255) & ~size_t(255); return p; };
  bf16_t* cur_bf = (bf16_t*)take((size_t)Mrows * DM * 2);
  bf16_t* hid_bf = (bf16_t*)take((size_t)Mrows * DM * 2);
  bf16_t* wq_bf  = (bf16_t*)take((size_t)DM * HD * 2);
  bf16_t* wkv_bf = (bf16_t*)take((size_t)DM * 2 * HD * 2);
  bf16_t* wo_bf  = (bf16_t*)take((size_t)HD * HD * 2);
  bf16_t* q_bf   = (bf16_t*)take((size_t)Mrows * HD * 2);
  bf16_t* kv_bf  = (bf16_t*)take((size_t)Mrows * 2 * HD * 2);
  bf16_t* ao_bf  = (bf16_t*)take((size_t)Mrows * HD * 2);

  auto cvt = [&](const float* src, bf16_t* dst, int n) {
    cvt_f32_bf16<<<(n + 255) / 256, 256, 0, stream>>>(src, dst, n);
  };
  cvt(current, cur_bf, Mrows * DM);
  cvt(hidden,  hid_bf, Mrows * DM);
  cvt(Wq,  wq_bf,  DM * HD);
  cvt(Wkv, wkv_bf, DM * 2 * HD);
  cvt(Wo,  wo_bf,  HD * HD);

  dim3 blk(256);
  // Q = current * Wq        (8192x512x512) -> bf16
  gemm_bf16<true><<<dim3(Mrows / 64, HD / 128), blk, 0, stream>>>(
      cur_bf, wq_bf, q_bf, Mrows, HD, DM);
  // KV = hidden * Wkv       (8192x1024x512) -> bf16
  gemm_bf16<true><<<dim3(Mrows / 64, (2 * HD) / 128), blk, 0, stream>>>(
      hid_bf, wkv_bf, kv_bf, Mrows, 2 * HD, DM);
  // flash attention: grid (B*H, N/128)
  flash_attn<<<dim3(B * 8, N / 128), blk, 0, stream>>>(q_bf, kv_bf, ao_bf);
  // out = attn_out * Wo     (8192x512x512) -> fp32
  gemm_bf16<false><<<dim3(Mrows / 64, HD / 128), blk, 0, stream>>>(
      ao_bf, wo_bf, out, Mrows, HD, DM);
}
